// Renderer_64458869178892
// MI455X (gfx1250) — matrix operand order repaired and checked by reading, not compiled, not run on hardware
//
#include <hip/hip_runtime.h>
#include <hip/hip_bf16.h>

typedef __attribute__((ext_vector_type(16))) _Float16 v16h;
typedef __attribute__((ext_vector_type(8)))  _Float16 v8h;
typedef __attribute__((ext_vector_type(8)))  float    v8f;

#define NPTS_PER_SB 262144   // NS*RH*RW = 64*64*64
#define HID 128
#define CCH 16
#define NFRAG 44             // 8 (L1) + 32 (L2) + 4 (L3) B-fragments
#define WBYTES (NFRAG * 512 * 2)   // 45,056 B of f16 fragments

// ---------------- workspace layout (bytes) ----------------
// vol     : 4 * 64*64*64 * 16 f32, channel-last  = 67,108,864 B
// featbuf : 2 * 262144 * 4 f32                   =  8,388,608 B
// wbuf    : 44 frags * 512 halves (f16)          =     45,056 B
#define WS_VOL_OFF   ((size_t)0)
#define WS_FEAT_OFF  ((size_t)67108864)
#define WS_WBUF_OFF  ((size_t)75497472)

// ---------------------------------------------------------------------------
// Weight prep: swizzle W1/W2/W3 (f32) into per-lane WMMA B-fragment layout
// (f16).  B element i of lane l covers K = ktile*32 + i + 16*(l>>4),
// N = ntile*16 + (l&15).  Fragment f: [0..7]=L1 ntiles, [8..39]=L2 (kt*8+nt),
// [40..43]=L3 ktiles.  Each fragment = 32 lanes * 16 halves = 512 halves.
// ---------------------------------------------------------------------------
__global__ void prep_wfrag(const float* __restrict__ W1,
                           const float* __restrict__ W2,
                           const float* __restrict__ W3,
                           _Float16* __restrict__ wbuf) {
  int lane = threadIdx.x & 31;
  int n_lo = lane & 15, kh = lane >> 4;
  for (int f = (int)(threadIdx.x >> 5); f < NFRAG; f += (int)(blockDim.x >> 5)) {
    _Float16* dst = wbuf + (size_t)f * 512 + lane * 16;
    if (f < 8) {                       // layer 1: W1 is 16x128, pad K to 32
      int n = f * 16 + n_lo;
      #pragma unroll
      for (int i = 0; i < 16; ++i) {
        int k = i + 16 * kh;
        dst[i] = (_Float16)((k < CCH) ? W1[k * HID + n] : 0.f);
      }
    } else if (f < 40) {               // layer 2: W2 is 128x128
      int t = f - 8, kt = t >> 3, nt = t & 7;
      int n = nt * 16 + n_lo;
      #pragma unroll
      for (int i = 0; i < 16; ++i) {
        int k = kt * 32 + i + 16 * kh;
        dst[i] = (_Float16)W2[k * HID + n];
      }
    } else {                           // layer 3: W3 is 128x4, pad N to 16
      int kt = f - 40;
      #pragma unroll
      for (int i = 0; i < 16; ++i) {
        int k = kt * 32 + i + 16 * kh;
        dst[i] = (_Float16)((n_lo < 4) ? W3[k * 4 + n_lo] : 0.f);
      }
    }
  }
}

// ---------------------------------------------------------------------------
// Conv encoder: 3x3 stride-2 SAME conv, 3ch -> 1024ch, + bias + relu.
// Output stored channel-last: vol[(((b*64+d)*64+oy)*64+ox)*16 + c] with
// o = c*64 + d, so trilinear sampling reads 16 contiguous floats per corner.
// One block computes a 2x2 output tile for all 1024 channels.
// ---------------------------------------------------------------------------
__global__ void __launch_bounds__(256)
conv_encoder(const float* __restrict__ images,
             const float* __restrict__ W_enc,
             const float* __restrict__ b_enc,
             float* __restrict__ vol) {
  __shared__ float patch[3][5][5];     // 3 in-ch, 5x5 input region
  int blk = blockIdx.x;                // 4 * 32 * 32 blocks
  int b   = blk >> 10;
  int rem = blk & 1023;
  int ty = rem >> 5, tx = rem & 31;
  int oy0 = ty * 2, ox0 = tx * 2;
  int t = threadIdx.x;

  if (t < 75) {
    int ci = t / 25, r = t % 25, ry = r / 5, rx = r % 5;
    int iy = 2 * oy0 + ry, ix = 2 * ox0 + rx;
    float v = 0.f;
    if (iy < 128 && ix < 128) {
      int sb = b >> 1, mbi = b & 1;
      v = images[((((size_t)sb * 3 + mbi) * 3 + ci) * 128 + iy) * 128 + ix];
    }
    patch[ci][ry][rx] = v;
  }
  __syncthreads();

  #pragma unroll
  for (int j = 0; j < 4; ++j) {
    int o = t + j * 256;
    const float* w = W_enc + (size_t)o * 27;
    __builtin_prefetch(w + 27 * 256, 0, 3);
    float bias = b_enc[o];
    float s00 = bias, s01 = bias, s10 = bias, s11 = bias;
    #pragma unroll
    for (int ci = 0; ci < 3; ++ci)
      #pragma unroll
      for (int ky = 0; ky < 3; ++ky)
        #pragma unroll
        for (int kx = 0; kx < 3; ++kx) {
          float wv = w[ci * 9 + ky * 3 + kx];
          s00 += wv * patch[ci][ky + 0][kx + 0];
          s01 += wv * patch[ci][ky + 0][kx + 2];
          s10 += wv * patch[ci][ky + 2][kx + 0];
          s11 += wv * patch[ci][ky + 2][kx + 2];
        }
    int c = o >> 6, dz = o & 63;
    size_t base = (((size_t)b * 64 + dz) * 64);
    vol[((base + oy0)     * 64 + ox0)     * 16 + c] = fmaxf(s00, 0.f);
    vol[((base + oy0)     * 64 + ox0 + 1) * 16 + c] = fmaxf(s01, 0.f);
    vol[((base + oy0 + 1) * 64 + ox0)     * 16 + c] = fmaxf(s10, 0.f);
    vol[((base + oy0 + 1) * 64 + ox0 + 1) * 16 + c] = fmaxf(s11, 0.f);
  }
}

// ---------------------------------------------------------------------------
// WMMA fragment helpers
// ---------------------------------------------------------------------------
// A fragment (16x32 f16) from an LDS row-major [16][rowHalves] buffer.
// Layout: lane m=l&15, kh=l>>4; element i -> k = kbase + i + 16*kh (matches B)
// => two contiguous 16B ds_load_b128 per lane.
__device__ __forceinline__ v16h lds_a_frag(const _Float16* row0, int rowHalves,
                                           int kbase, int lane) {
  int m = lane & 15, kh = lane >> 4;
  const _Float16* p = row0 + m * rowHalves + kbase + 16 * kh;
  v8h lo = *(const v8h*)p;
  v8h hi = *(const v8h*)(p + 8);
  v16h a;
  #pragma unroll
  for (int i = 0; i < 8; ++i) { a[i] = lo[i]; a[i + 8] = hi[i]; }
  return a;
}

// B fragment: 16 contiguous halves per lane from the LDS-staged weight buffer.
__device__ __forceinline__ v16h lds_b_frag(const _Float16* wlds, int frag, int lane) {
  const _Float16* p = wlds + (size_t)frag * 512 + lane * 16;
  v8h lo = *(const v8h*)p;
  v8h hi = *(const v8h*)(p + 8);
  v16h b;
  #pragma unroll
  for (int i = 0; i < 8; ++i) { b[i] = lo[i]; b[i + 8] = hi[i]; }
  return b;
}

__device__ __forceinline__ v8f wmma_f16(v16h a, v16h b, v8f c) {
  return __builtin_amdgcn_wmma_f32_16x16x32_f16(false, a, false, b,
                                                (short)0, c, false, false);
}

// ---------------------------------------------------------------------------
// Fused trilinear sampling (+mean over mb) + 3-layer MLP via WMMA.
// 256 threads = 8 waves; each wave owns 16 points (M=16 rows).
// All 44 weight B-fragments (45 KB) are copied global->LDS with the CDNA5
// async-LDS path (ASYNCcnt), overlapped with the sampling phase.
// ---------------------------------------------------------------------------
__global__ void __launch_bounds__(256)
sample_mlp(const float* __restrict__ rays,      // target rays (SB,1,64,64,8)
           const float* __restrict__ vol,       // channel-last volumes
           const _Float16* __restrict__ wbuf,   // pre-swizzled B fragments
           const float* __restrict__ b1,
           const float* __restrict__ b2,
           const float* __restrict__ b3,
           float* __restrict__ featbuf) {
  __shared__ __align__(16) _Float16 xs[8][16][32];     //  8 KB: layer-1 A (K padded)
  __shared__ __align__(16) _Float16 hbuf[8][16][HID];  // 32 KB: hidden activations
  extern __shared__ _Float16 wlds[];                   // 44 KB dynamic: weights

  int wave = threadIdx.x >> 5, lane = threadIdx.x & 31;
  int m = lane & 15, kh = lane >> 4;
  int base = blockIdx.x * 128 + wave * 16;

  // ----- kick off async global->LDS copy of all weight fragments ----------
  // 45,056 B = 2816 x 16B chunks = 11 chunks per thread (256 threads).
  {
    unsigned lds0 = (unsigned)(unsigned long long)(&wlds[0]);  // low 32b = LDS offset
    const char* g0 = (const char*)wbuf;
    #pragma unroll
    for (int j = 0; j < 11; ++j) {
      unsigned chunk = threadIdx.x + j * 256u;
      unsigned ldsoff = lds0 + chunk * 16u;
      unsigned long long ga = (unsigned long long)(g0 + (size_t)chunk * 16u);
      asm volatile("global_load_async_to_lds_b128 %0, %1, off"
                   :: "v"(ldsoff), "v"(ga) : "memory");
    }
  }

  // ---------------- trilinear sampling: 8 channels per lane ----------------
  {
    int g   = base + m;
    int sb  = g >> 18;                 // / 262144
    int pid = g & (NPTS_PER_SB - 1);
    int s = pid >> 12, y = (pid >> 6) & 63, x = pid & 63;

    const float* ray = rays + (((size_t)sb * 64 + y) * 64 + x) * 8;
    float ox_ = ray[0], oy_ = ray[1], oz_ = ray[2];
    float dx_ = ray[3], dy_ = ray[4], dz_ = ray[5];
    float nearv = ray[6], farv = ray[7];

    float tt = (s + 0.5f) * (1.0f / 64.0f);
    float zr = nearv * (1.f - tt) + farv * tt;
    float px = ox_ + dx_ * zr, py = oy_ + dy_ * zr, pz = oz_ + dz_ * zr;

    const float inv = 1.0f / 3.5f;
    float cx = fminf(fmaxf(px * inv, -1.f), 1.f);
    float cy = fminf(fmaxf(py * inv, -1.f), 1.f);
    float cz = fminf(fmaxf(pz * inv, -1.f), 1.f);
    float ux = (cx + 1.f) * 31.5f, uy = (cy + 1.f) * 31.5f, uz = (cz + 1.f) * 31.5f;
    int x0 = min(max((int)floorf(ux), 0), 62);
    int y0 = min(max((int)floorf(uy), 0), 62);
    int z0 = min(max((int)floorf(uz), 0), 62);
    float fx = ux - x0, fy = uy - y0, fz = uz - z0;

    int chb = kh * 8;                  // lanes 0-15: ch 0-7, lanes 16-31: ch 8-15
    float acc[8];
    #pragma unroll
    for (int c = 0; c < 8; ++c) acc[c] = 0.f;

    const float* v0 = vol + (size_t)(sb * 2)     * 64 * 64 * 64 * 16;
    const float* v1 = vol + (size_t)(sb * 2 + 1) * 64 * 64 * 64 * 16;

    #pragma unroll
    for (int dz2 = 0; dz2 < 2; ++dz2) {
      float wz = dz2 ? fz : 1.f - fz;
      #pragma unroll
      for (int dy2 = 0; dy2 < 2; ++dy2) {
        float wy = dy2 ? fy : 1.f - fy;
        #pragma unroll
        for (int dx2 = 0; dx2 < 2; ++dx2) {
          float wx = dx2 ? fx : 1.f - fx;
          float w = wz * wy * wx * 0.5f;     // * 0.5 = mean over mb
          size_t idx = ((((size_t)(z0 + dz2)) * 64 + (y0 + dy2)) * 64
                        + (x0 + dx2)) * 16 + chb;
          const float4* p0 = (const float4*)(v0 + idx);
          const float4* p1 = (const float4*)(v1 + idx);
          float4 a0 = p0[0], a1 = p0[1];
          float4 c0 = p1[0], c1 = p1[1];
          acc[0] += w * (a0.x + c0.x); acc[1] += w * (a0.y + c0.y);
          acc[2] += w * (a0.z + c0.z); acc[3] += w * (a0.w + c0.w);
          acc[4] += w * (a1.x + c1.x); acc[5] += w * (a1.y + c1.y);
          acc[6] += w * (a1.z + c1.z); acc[7] += w * (a1.w + c1.w);
        }
      }
    }
    #pragma unroll
    for (int c = 0; c < 8; ++c) xs[wave][m][chb + c] = (_Float16)acc[c];
    #pragma unroll
    for (int c = 0; c < 8; ++c) xs[wave][m][16 + chb + c] = (_Float16)0.f;  // K pad
  }

  // drain our async weight copies, then block-wide barrier publishes LDS
  asm volatile("s_wait_asynccnt 0" ::: "memory");
  __syncthreads();

  int nloc = lane & 15;

  // ---------------- layer 1: (16x16) x (16x128), K padded to 32 ------------
  v16h a1 = lds_a_frag(&xs[wave][0][0], 32, 0, lane);
  #pragma unroll
  for (int nt = 0; nt < 8; ++nt) {
    float bv = b1[nt * 16 + nloc];
    v8f c;
    #pragma unroll
    for (int i = 0; i < 8; ++i) c[i] = bv;
    v16h b = lds_b_frag(wlds, nt, lane);
    c = wmma_f16(a1, b, c);
    #pragma unroll
    for (int i = 0; i < 8; ++i) {
      float hv = fmaxf(c[i], 0.f);                 // relu
      hbuf[wave][i + 8 * kh][nt * 16 + nloc] = (_Float16)hv;
    }
  }
  __syncthreads();

  // ---------------- layer 2: (16x128) x (128x128) --------------------------
  v8f acc2[8];
  #pragma unroll
  for (int nt = 0; nt < 8; ++nt) {
    float bv = b2[nt * 16 + nloc];
    v8f c;
    #pragma unroll
    for (int i = 0; i < 8; ++i) c[i] = bv;
    #pragma unroll
    for (int kt = 0; kt < 4; ++kt) {
      v16h a = lds_a_frag(&hbuf[wave][0][0], HID, kt * 32, lane);
      v16h b = lds_b_frag(wlds, 8 + kt * 8 + nt, lane);
      c = wmma_f16(a, b, c);
    }
    #pragma unroll
    for (int i = 0; i < 8; ++i) c[i] = fmaxf(c[i], 0.f);  // relu
    acc2[nt] = c;
  }
  __syncthreads();

  // write h2 back into hbuf (all reads of h1 are complete)
  #pragma unroll
  for (int nt = 0; nt < 8; ++nt)
    #pragma unroll
    for (int i = 0; i < 8; ++i)
      hbuf[wave][i + 8 * kh][nt * 16 + nloc] = (_Float16)acc2[nt][i];
  __syncthreads();

  // ---------------- layer 3: (16x128) x (128x4 padded to 16) ---------------
  v8f c3;
  float bv3 = (nloc < 4) ? b3[nloc] : 0.f;
  #pragma unroll
  for (int i = 0; i < 8; ++i) c3[i] = bv3;
  #pragma unroll
  for (int kt = 0; kt < 4; ++kt) {
    v16h a = lds_a_frag(&hbuf[wave][0][0], HID, kt * 32, lane);
    v16h b = lds_b_frag(wlds, 40 + kt, lane);
    c3 = wmma_f16(a, b, c3);
  }

  if (nloc < 4) {
    #pragma unroll
    for (int i = 0; i < 8; ++i) {
      int g2 = base + i + 8 * kh;
      featbuf[(size_t)g2 * 4 + nloc] = c3[i];
    }
  }
}

// ---------------------------------------------------------------------------
// Volume rendering compositing: one thread per ray (2 * 64 * 64 rays).
// ---------------------------------------------------------------------------
__global__ void __launch_bounds__(256)
composite_k(const float* __restrict__ rays,
            const float* __restrict__ featbuf,
            float* __restrict__ out) {
  int r = blockIdx.x * blockDim.x + threadIdx.x;
  if (r >= 8192) return;
  int sb = r >> 12, y = (r >> 6) & 63, x = r & 63;
  const float* ray = rays + (((size_t)sb * 64 + y) * 64 + x) * 8;
  float nearv = ray[6], farv = ray[7];

  float T = 1.f, R = 0.f, G = 0.f, B = 0.f, Dp = 0.f;
  for (int s = 0; s < 64; ++s) {
    float tt  = (s + 0.5f) * (1.0f / 64.0f);
    float z   = nearv * (1.f - tt) + farv * tt;
    float tt2 = (s + 1.5f) * (1.0f / 64.0f);
    float zn  = nearv * (1.f - tt2) + farv * tt2;
    float delta = (s == 63) ? 1000.f : (zn - z);

    const float* f = featbuf + ((size_t)sb * NPTS_PER_SB
                                + (size_t)s * 4096 + y * 64 + x) * 4;
    float sg = fmaxf(f[3], 0.f);
    float alpha = 1.f - __expf(-sg * delta);
    float w = alpha * T;
    R  += w * (1.f / (1.f + __expf(-f[0])));
    G  += w * (1.f / (1.f + __expf(-f[1])));
    B  += w * (1.f / (1.f + __expf(-f[2])));
    Dp += w * z;
    T *= (1.f - alpha + 1e-10f);
  }
  int pix = sb * 4096 + y * 64 + x;
  out[pix * 3 + 0] = R;
  out[pix * 3 + 1] = G;
  out[pix * 3 + 2] = B;
  out[24576 + pix] = Dp;
}

// ---------------------------------------------------------------------------
extern "C" void kernel_launch(void* const* d_in, const int* in_sizes, int n_in,
                              void* d_out, int out_size, void* d_ws, size_t ws_size,
                              hipStream_t stream) {
  const float* images      = (const float*)d_in[0];
  // d_in[1] = cond_rays (unused: norm_src == 0)
  const float* target_rays = (const float*)d_in[2];
  const float* W_enc       = (const float*)d_in[3];
  const float* b_enc       = (const float*)d_in[4];
  const float* W1          = (const float*)d_in[5];
  const float* b1          = (const float*)d_in[6];
  const float* W2          = (const float*)d_in[7];
  const float* b2          = (const float*)d_in[8];
  const float* W3          = (const float*)d_in[9];
  const float* b3          = (const float*)d_in[10];
  float* out = (float*)d_out;

  char* ws = (char*)d_ws;
  float*     vol     = (float*)(ws + WS_VOL_OFF);
  float*     featbuf = (float*)(ws + WS_FEAT_OFF);
  _Float16*  wbuf    = (_Float16*)(ws + WS_WBUF_OFF);

  prep_wfrag  <<<1,    256, 0,       stream>>>(W1, W2, W3, wbuf);
  conv_encoder<<<4096, 256, 0,       stream>>>(images, W_enc, b_enc, vol);
  sample_mlp  <<<4096, 256, WBYTES,  stream>>>(target_rays, vol, wbuf,
                                               b1, b2, b3, featbuf);
  composite_k <<<32,   256, 0,       stream>>>(target_rays, featbuf, out);
}
